// Encoder_39591008534759
// MI455X (gfx1250) — matrix operand order repaired and checked by reading, not compiled, not run on hardware
//
#include <hip/hip_runtime.h>
#include <hip/hip_bf16.h>

// ---------------------------------------------------------------------------
// 2-layer GCN for gfx1250 (MI455X), wave32 + WMMA bf16.
//   h1 = relu( scatter(norm * (x @W1)[src] -> dst) + b1 )
//   out= relu( scatter(norm * (h1@W2)[src] -> dst) + b2 )
// ---------------------------------------------------------------------------

typedef __attribute__((ext_vector_type(16))) __bf16 v16bf;
typedef __attribute__((ext_vector_type(8)))  float  v8f;

union FragBF {
    unsigned short u[16];
    uint4          q[2];
    v16bf          v;
};

__device__ __forceinline__ unsigned short f2bf(float f) {
    __bf16 h = (__bf16)f;   // native fp32->bf16 (RNE) convert
    return __builtin_bit_cast(unsigned short, h);
}

// ---------------------------------------------------------------------------
// GEMM: out[M x OUT] = A[M x 128] * W[128 x OUT], bf16 WMMA, fp32 accumulate.
// One 16x16 tile per wave. Both operands staged in LDS pre-swizzled to
// fragment order so each lane's 16 elements are 2 contiguous ds_load_b128:
//   W: lds[ ((k>>4)*OUT + n)*16 + (k&15) ]                (B-frag order)
//   A: lds[ ((kk*2+half)*16 + row)*16 + e ]               (A-frag order)
//      where k = kk*32+rem, half=rem>>4, e=rem&15
// ---------------------------------------------------------------------------
template <int OUT_DIM>
__global__ __launch_bounds__(32 * (OUT_DIM / 16))
void gemm_bf16_wmma(const float* __restrict__ A, const float* __restrict__ W,
                    float* __restrict__ out, int n_rows, int m_tiles) {
    constexpr int NT = 32 * (OUT_DIM / 16);   // threads per block
    __shared__ unsigned short ldsW[128 * OUT_DIM];
    __shared__ unsigned short ldsA[16 * 128];

    const int tid = threadIdx.x;

    // stage + swizzle weights once per block (L2-hot; 32KB/16KB of 320KB LDS)
    for (int i = tid; i < 128 * OUT_DIM; i += NT) {
        const int k = i / OUT_DIM;            // power-of-two -> shifts
        const int n = i % OUT_DIM;
        ldsW[((k >> 4) * OUT_DIM + n) * 16 + (k & 15)] = f2bf(W[i]);
    }

    const int wave  = tid >> 5;
    const int lane  = tid & 31;
    const int lrow  = lane & 15;   // A row offset / B-D column offset
    const int lhalf = lane >> 4;   // K-half (A/B), M-half (C/D)
    const int nBase = wave * 16;   // one 16-wide N tile per wave

    for (int mt = blockIdx.x; mt < m_tiles; mt += gridDim.x) {
        const int mBase = mt * 16;

        // prefetch next tile's A rows (global_prefetch_b8)
        {
            const int nb = mBase + (int)gridDim.x * 16;
            if (nb + 16 <= n_rows)
                __builtin_prefetch(A + (size_t)nb * 128 + tid * 8, 0, 3);
        }

        __syncthreads();   // prior iteration's ldsA readers done (also ldsW on iter 0)
        // stage + swizzle this block's 16x128 A tile (converted once per block)
        for (int i = tid; i < 16 * 128; i += NT) {
            const int row  = i >> 7;
            const int k    = i & 127;
            const int kk   = k >> 5;
            const int rem  = k & 31;
            const int half = rem >> 4;
            const int e    = rem & 15;
            int r = mBase + row;
            if (r >= n_rows) r = n_rows - 1;
            ldsA[((kk * 2 + half) * 16 + row) * 16 + e] = f2bf(A[(size_t)r * 128 + k]);
        }
        __syncthreads();

        v8f acc = {};
#pragma unroll
        for (int kk = 0; kk < 4; ++kk) {
            FragBF a, b;
            const unsigned short* ap = &ldsA[((kk * 2 + lhalf) * 16 + lrow) * 16];
            a.q[0] = *(const uint4*)(ap + 0);
            a.q[1] = *(const uint4*)(ap + 8);
            const unsigned short* bp =
                &ldsW[((kk * 2 + lhalf) * OUT_DIM + (nBase + lrow)) * 16];
            b.q[0] = *(const uint4*)(bp + 0);
            b.q[1] = *(const uint4*)(bp + 8);
            acc = __builtin_amdgcn_wmma_f32_16x16x32_bf16(
                false, a.v, false, b.v, (short)0, acc, false, false);
        }

        // store D: lane col = nBase+lrow, row = mBase + lhalf*8 + r
        const int mrow = mBase + lhalf * 8;
        float* op = out + (size_t)mrow * OUT_DIM + nBase + lrow;
        if (mBase + 16 <= n_rows) {     // tile-uniform fast path (always taken here)
#pragma unroll
            for (int r = 0; r < 8; ++r) op[(size_t)r * OUT_DIM] = acc[r];
        } else {
#pragma unroll
            for (int r = 0; r < 8; ++r)
                if (mrow + r < n_rows) op[(size_t)r * OUT_DIM] = acc[r];
        }
    }
}

// ---------------------------------------------------------------------------
// Degree / normalization (integer atomics -> deterministic degrees)
// ---------------------------------------------------------------------------
__global__ void deg_init(int* __restrict__ deg, int n) {
    int i = blockIdx.x * blockDim.x + threadIdx.x;
    if (i < n) deg[i] = 1;  // self loop
}

__global__ void deg_edges(const int* __restrict__ ei, int* __restrict__ deg, int E) {
    int e = blockIdx.x * blockDim.x + threadIdx.x;
    if (e < E) atomicAdd(&deg[ei[E + e]], 1);  // dst row of edge_index
}

__global__ void make_dinv(const int* __restrict__ deg, float* __restrict__ dinv, int n) {
    int i = blockIdx.x * blockDim.x + threadIdx.x;
    if (i < n) dinv[i] = rsqrtf((float)deg[i]);
}

// ---------------------------------------------------------------------------
// Aggregation: agg = dinv[i]^2 * xw[i]  (self loop), then edge scatter-adds.
// ---------------------------------------------------------------------------
template <int D>
__global__ void agg_self(const float4* __restrict__ xw, const float* __restrict__ dinv,
                         float4* __restrict__ agg, int n) {
    constexpr int D4 = D / 4;
    int idx = blockIdx.x * blockDim.x + threadIdx.x;
    if (idx < n * D4) {
        const int i = idx / D4;
        float s = dinv[i];
        s *= s;
        float4 v = xw[idx];
        agg[idx] = make_float4(v.x * s, v.y * s, v.z * s, v.w * s);
    }
}

template <int D>
__global__ void agg_edges(const int* __restrict__ ei, const float* __restrict__ xw,
                          const float* __restrict__ dinv, float* __restrict__ agg, int E) {
    int idx = blockIdx.x * blockDim.x + threadIdx.x;
    const int e = idx / D;
    const int f = idx % D;
    if (e < E) {
        const int s = ei[e];
        const int d = ei[E + e];
        const float w = dinv[s] * dinv[d];
        unsafeAtomicAdd(&agg[(size_t)d * D + f], w * xw[(size_t)s * D + f]);
    }
}

template <int D>
__global__ void bias_relu(const float4* __restrict__ agg, const float4* __restrict__ b,
                          float4* __restrict__ h, int n) {
    constexpr int D4 = D / 4;
    int idx = blockIdx.x * blockDim.x + threadIdx.x;
    if (idx < n * D4) {
        const float4 v  = agg[idx];
        const float4 bb = b[idx % D4];
        h[idx] = make_float4(fmaxf(v.x + bb.x, 0.0f), fmaxf(v.y + bb.y, 0.0f),
                             fmaxf(v.z + bb.z, 0.0f), fmaxf(v.w + bb.w, 0.0f));
    }
}

// ---------------------------------------------------------------------------
static inline size_t align256(size_t x) { return (x + 255) & ~(size_t)255; }
static inline int imin(int a, int b) { return a < b ? a : b; }

extern "C" void kernel_launch(void* const* d_in, const int* in_sizes, int n_in,
                              void* d_out, int out_size, void* d_ws, size_t ws_size,
                              hipStream_t stream) {
    const float* x  = (const float*)d_in[0];
    const int*   ei = (const int*)  d_in[1];
    const float* W1 = (const float*)d_in[2];
    const float* b1 = (const float*)d_in[3];
    const float* W2 = (const float*)d_in[4];
    const float* b2 = (const float*)d_in[5];
    float*       out = (float*)d_out;

    const int n = in_sizes[0] / 128;   // 100000
    const int E = in_sizes[1] / 2;     // 800000

    // workspace layout
    char*  ws  = (char*)d_ws;
    size_t off = 0;
    int*   deg  = (int*)(ws + off);   off += align256((size_t)n * 4);
    float* dinv = (float*)(ws + off); off += align256((size_t)n * 4);
    float* xw1  = (float*)(ws + off); off += align256((size_t)n * 128 * 4);
    float* agg1 = (float*)(ws + off); off += align256((size_t)n * 128 * 4);
    float* h1   = xw1;                               // reuse: xw1 dead after agg
    float* xw2  = agg1;                              // reuse: agg1 dead after relu
    float* agg2 = (float*)((char*)agg1 + (size_t)n * 64 * 4);

    const int T = 256;
    const int mTiles = (n + 15) / 16;
    const int gGrid  = imin(mTiles, 1024);   // amortize W staging over ~6 tiles

    // normalization
    deg_init <<<(n + T - 1) / T, T, 0, stream>>>(deg, n);
    deg_edges<<<(E + T - 1) / T, T, 0, stream>>>(ei, deg, E);
    make_dinv<<<(n + T - 1) / T, T, 0, stream>>>(deg, dinv, n);

    // ----- layer 1: D = 128 -----
    gemm_bf16_wmma<128><<<gGrid, 256, 0, stream>>>(x, W1, xw1, n, mTiles);
    agg_self<128><<<((size_t)n * 32 + T - 1) / T, T, 0, stream>>>(
        (const float4*)xw1, dinv, (float4*)agg1, n);
    agg_edges<128><<<((size_t)E * 128 + T - 1) / T, T, 0, stream>>>(ei, xw1, dinv, agg1, E);
    bias_relu<128><<<((size_t)n * 32 + T - 1) / T, T, 0, stream>>>(
        (const float4*)agg1, (const float4*)b1, (float4*)h1, n);

    // ----- layer 2: D = 64 -----
    gemm_bf16_wmma<64><<<gGrid, 128, 0, stream>>>(h1, W2, xw2, n, mTiles);
    agg_self<64><<<((size_t)n * 16 + T - 1) / T, T, 0, stream>>>(
        (const float4*)xw2, dinv, (float4*)agg2, n);
    agg_edges<64><<<((size_t)E * 64 + T - 1) / T, T, 0, stream>>>(ei, xw2, dinv, agg2, E);
    bias_relu<64><<<((size_t)n * 16 + T - 1) / T, T, 0, stream>>>(
        (const float4*)agg2, (const float4*)b2, (float4*)out, n);
}
